// AngularMarginLoss_59176059405076
// MI455X (gfx1250) — matrix operand order repaired and checked by reading, not compiled, not run on hardware
//
#include <hip/hip_runtime.h>
#include <hip/hip_bf16.h>

// CDNA5 / gfx1250, wave32. WMMA bf16 16x16x32.
typedef __attribute__((ext_vector_type(16))) __bf16 v16bf;
typedef __attribute__((ext_vector_type(8)))  float  v8f;

#define B_N   1024
#define E_K   256
#define C_N   100000
#define SCALE_F 30.0f
#define M2_F  0.5f
#define EPS_F 1e-6f

// Map float -> monotonically ordered uint (for packed argmax keys).
__device__ __forceinline__ unsigned int float_to_ordered(float f) {
    unsigned int u = __float_as_uint(f);
    unsigned int mask = (unsigned int)((int)u >> 31) | 0x80000000u;
    return u ^ mask;
}

__device__ __forceinline__ unsigned long long shfl_xor_u64(unsigned long long v, int m) {
    unsigned int lo = __shfl_xor((unsigned int)v, m, 32);
    unsigned int hi = __shfl_xor((unsigned int)(v >> 32), m, 32);
    return ((unsigned long long)hi << 32) | lo;
}

// ---------------------------------------------------------------------------
// K1: L2-normalize input rows -> bf16 xn (workspace); passthrough raw inputs
//     to d_out; init per-sample accumulators. 8 rows per 256-thread block.
// ---------------------------------------------------------------------------
__global__ void k_norm_x(const float* __restrict__ x, float* __restrict__ out_x,
                         __bf16* __restrict__ xn,
                         unsigned long long* __restrict__ g_max,
                         float* __restrict__ g_sum, float* __restrict__ g_cost) {
    int tid = threadIdx.x;
    int gid = blockIdx.x * 256 + tid;
    if (gid < B_N) { g_max[gid] = 0ull; g_sum[gid] = 0.0f; g_cost[gid] = 0.0f; }
    int wave = tid >> 5, lane = tid & 31;
    int row = blockIdx.x * 8 + wave;            // 128 blocks * 8 = 1024 rows exactly
    const float* p = x + (size_t)row * E_K + lane * 8;
    float v[8];
    float ss = 0.0f;
#pragma unroll
    for (int j = 0; j < 8; ++j) { v[j] = p[j]; ss += v[j] * v[j]; }
#pragma unroll
    for (int m = 16; m >= 1; m >>= 1) ss += __shfl_xor(ss, m, 32);
    float inv = rsqrtf(ss);
    float* po = out_x + (size_t)row * E_K + lane * 8;
    __bf16* pn = xn + (size_t)row * E_K + lane * 8;
#pragma unroll
    for (int j = 0; j < 8; ++j) { po[j] = v[j]; pn[j] = (__bf16)(v[j] * inv); }
}

// ---------------------------------------------------------------------------
// K2: reciprocal L2 norm per weight row. 8 rows per block (wave per row).
// ---------------------------------------------------------------------------
__global__ void k_norm_w(const float* __restrict__ W, float* __restrict__ winv) {
    int tid = threadIdx.x;
    int wave = tid >> 5, lane = tid & 31;
    int row = blockIdx.x * 8 + wave;            // 12500 blocks * 8 = 100000 exactly
    const float* p = W + (size_t)row * E_K + lane * 8;
    float ss = 0.0f;
#pragma unroll
    for (int j = 0; j < 8; ++j) { float w = p[j]; ss += w * w; }
#pragma unroll
    for (int m = 16; m >= 1; m >>= 1) ss += __shfl_xor(ss, m, 32);
    if (lane == 0) winv[row] = rsqrtf(ss);
}

// ---------------------------------------------------------------------------
// K3: fused GEMM (WMMA bf16) + sum-exp / argmax / target-cos epilogue.
// Each wave owns a 32-class strip (2 x 16-class WMMA column tiles), keeps the
// normalized bf16 B fragments for K=256 resident in VGPRs (W read ONCE from
// HBM), and streams all 64 row-blocks of xn (L2-resident) through the matrix
// pipes. Per-sample partials accumulate via LDS atomics; one global atomic
// flush per block.
// ---------------------------------------------------------------------------
__global__ __launch_bounds__(256, 1) void k_main(
    const float* __restrict__ W, const float* __restrict__ winv,
    const __bf16* __restrict__ xn, const int* __restrict__ targets,
    unsigned long long* __restrict__ g_max, float* __restrict__ g_sum,
    float* __restrict__ g_cost)
{
    __shared__ float              lds_sum[B_N];   // 4 KB
    __shared__ unsigned long long lds_max[B_N];   // 8 KB
    __shared__ int                lds_tgt[B_N];   // 4 KB

    int tid = threadIdx.x;
#pragma unroll
    for (int i = 0; i < 4; ++i) {
        int idx = tid + i * 256;
        lds_sum[idx] = 0.0f;
        lds_max[idx] = 0ull;
        lds_tgt[idx] = targets[idx];
    }
    __syncthreads();

    int wave = tid >> 5, lane = tid & 31;
    int strip = blockIdx.x * 8 + wave;           // 32 classes per strip; 3125 strips
    if (strip * 32 < C_N) {
        int n     = lane & 15;                   // column within 16-wide tile
        int khalf = lane >> 4;                   // lane-half selects K sub-range

        // --- load + normalize register-resident B fragments (K = 256) ---
        // B k order chosen here (32x16 bf16, contiguous, same as A): lanes 0-15 hold K=0..15 (V_i: K=2i,2i+1),
        // lanes 16-31 hold K=16..31.
        v16bf bfrag[2][8];
#pragma unroll
        for (int t = 0; t < 2; ++t) {
            int c = strip * 32 + t * 16 + n;
            float inv = winv[c];
            const float* wrow = W + (size_t)c * E_K;
            __builtin_prefetch(wrow, 0, 1);
#pragma unroll
            for (int kc = 0; kc < 8; ++kc) {
#pragma unroll
                for (int vg = 0; vg < 8; ++vg) {
                    int k = kc * 32 + khalf * 16 + vg * 2;
                    bfrag[t][kc][2 * vg]     = (__bf16)(wrow[k] * inv);
                    bfrag[t][kc][2 * vg + 1] = (__bf16)(wrow[k + 1] * inv);
                }
            }
        }
        int c0 = strip * 32 + n;
        int c1 = c0 + 16;

        for (int rb = 0; rb < B_N / 16; ++rb) {
            // A k order chosen here (16x32 bf16, contiguous, same as B): lane holds row M = lane&15; lane-half
            // picks K sub-blocks 0..15 vs 16..31.
            int mrow = rb * 16 + n;
            const __bf16* arow = xn + (size_t)mrow * E_K;
            v8f acc0 = {}; v8f acc1 = {};
#pragma unroll
            for (int kc = 0; kc < 8; ++kc) {
                v16bf a;
#pragma unroll
                for (int vg = 0; vg < 8; ++vg) {
                    int k = kc * 32 + khalf * 16 + vg * 2;
                    a[2 * vg]     = arow[k];
                    a[2 * vg + 1] = arow[k + 1];
                }
                acc0 = __builtin_amdgcn_wmma_f32_16x16x32_bf16(
                           false, a, false, bfrag[0][kc], (short)0, acc0, false, false);
                acc1 = __builtin_amdgcn_wmma_f32_16x16x32_bf16(
                           false, a, false, bfrag[1][kc], (short)0, acc1, false, false);
            }
            // D layout: lane holds column n; VGPR r holds row r + 8*khalf.
#pragma unroll
            for (int r = 0; r < 8; ++r) {
                int brow = rb * 16 + khalf * 8 + r;
                float v0 = fminf(fmaxf(acc0[r], -1.0f), 1.0f);
                float v1 = fminf(fmaxf(acc1[r], -1.0f), 1.0f);
                int tgt = lds_tgt[brow];
                if (c0 == tgt) g_cost[brow] = v0;   // unique writer per sample
                if (c1 == tgt) g_cost[brow] = v1;
                float es = __expf(SCALE_F * v0) + __expf(SCALE_F * v1);
                unsigned long long k0 =
                    ((unsigned long long)float_to_ordered(v0) << 32) | (unsigned int)(~c0);
                unsigned long long k1 =
                    ((unsigned long long)float_to_ordered(v1) << 32) | (unsigned int)(~c1);
                unsigned long long mk = (k0 > k1) ? k0 : k1;
                // reduce across the 16 lanes sharing this row (stays in half)
#pragma unroll
                for (int m = 1; m <= 8; m <<= 1) {
                    es += __shfl_xor(es, m, 32);
                    unsigned long long o = shfl_xor_u64(mk, m);
                    if (o > mk) mk = o;
                }
                if (n == 0) {
                    atomicAdd(&lds_sum[brow], es);   // ds_add_f32
                    atomicMax(&lds_max[brow], mk);   // ds_max_u64
                }
            }
        }
    }
    __syncthreads();
    // flush block partials (one global atomic pair per sample per block)
#pragma unroll
    for (int i = 0; i < 4; ++i) {
        int idx = tid + i * 256;
        float s = lds_sum[idx];
        if (s != 0.0f) atomicAdd(&g_sum[idx], s);
        unsigned long long mk = lds_max[idx];
        if (mk != 0ull) atomicMax(&g_max[idx], mk);
    }
}

// ---------------------------------------------------------------------------
// K4: per-sample margin transform + loss mean + preds. Single block.
// numerator = 30*cos(arccos(c)+0.5) = 30*(c*cos0.5 - sqrt(1-c^2)*sin0.5)
// ---------------------------------------------------------------------------
__global__ void k_final(const unsigned long long* __restrict__ g_max,
                        const float* __restrict__ g_sum,
                        const float* __restrict__ g_cost,
                        float* __restrict__ out) {
    __shared__ float red[256];
    int tid = threadIdx.x;
    const float cm = cosf(M2_F), sm = sinf(M2_F);
    float part = 0.0f;
#pragma unroll
    for (int i = 0; i < 4; ++i) {
        int b = tid + i * 256;
        float ct  = g_cost[b];
        float num = SCALE_F * (ct * cm - sqrtf(fmaxf(0.0f, 1.0f - ct * ct)) * sm);
        float sum_excl = g_sum[b] - __expf(SCALE_F * ct);
        float denom = __expf(num) + sum_excl;
        part += num - __logf(denom + EPS_F);
        unsigned int pred = ~(unsigned int)(g_max[b] & 0xffffffffu);
        out[B_N * E_K + b] = (float)pred;
    }
    red[tid] = part;
    __syncthreads();
    for (int s = 128; s > 0; s >>= 1) {
        if (tid < s) red[tid] += red[tid + s];
        __syncthreads();
    }
    if (tid == 0) out[B_N * E_K + B_N] = -red[0] / (float)B_N;
}

// ---------------------------------------------------------------------------
extern "C" void kernel_launch(void* const* d_in, const int* in_sizes, int n_in,
                              void* d_out, int out_size, void* d_ws, size_t ws_size,
                              hipStream_t stream) {
    const float* x       = (const float*)d_in[0];
    const int*   targets = (const int*)d_in[1];
    const float* W       = (const float*)d_in[2];
    float* out = (float*)d_out;
    char*  ws  = (char*)d_ws;

    // workspace layout (all 8B-aligned): total ~919 KB
    __bf16*             xn     = (__bf16*)ws;                          // 524288 B
    float*              winv   = (float*)(ws + 524288);                // 400000 B
    unsigned long long* g_max  = (unsigned long long*)(ws + 924288);   //   8192 B
    float*              g_sum  = (float*)(ws + 932480);                //   4096 B
    float*              g_cost = (float*)(ws + 936576);                //   4096 B

    hipLaunchKernelGGL(k_norm_x, dim3(128),   dim3(256), 0, stream,
                       x, out, xn, g_max, g_sum, g_cost);
    hipLaunchKernelGGL(k_norm_w, dim3(12500), dim3(256), 0, stream, W, winv);
    hipLaunchKernelGGL(k_main,   dim3(391),   dim3(256), 0, stream,
                       W, winv, xn, targets, g_max, g_sum, g_cost);
    hipLaunchKernelGGL(k_final,  dim3(1),     dim3(256), 0, stream,
                       g_max, g_sum, g_cost, out);
}
